// GATLayer_2345052144201
// MI455X (gfx1250) — matrix operand order repaired and checked by reading, not compiled, not run on hardware
//
#include <hip/hip_runtime.h>

// GAT layer for MI455X (gfx1250, wave32, WMMA).
// VALU (softmax exp/mask math over 151M (i,j,h) pairs) is the binding
// constraint, not the 151MB adj stream. Single adj pass, no max-shift softmax
// (|e|<~10 so f32 exp2 is safe), denominator computed by a 5th WMMA against an
// all-ones bf16 B fragment (row sums land in C layout exactly where each lane
// needs them). All f32<->bf16 conversions are native __bf16 casts so the
// backend emits hardware cvt ops instead of integer round-and-shift chains.

#define Nn    6144
#define INF_  256
#define OUTF  256
#define HEADS 4
#define FPH   64
#define NEG_SLOPE 0.2f
#define L2E   1.44269504088896340736f   // log2(e)

typedef __attribute__((ext_vector_type(8)))  float        v8f;
typedef __attribute__((ext_vector_type(16))) __bf16       v16bf;
typedef __attribute__((ext_vector_type(8)))  __bf16       v8bf;
typedef __attribute__((ext_vector_type(8)))  unsigned int v8u;

__device__ __forceinline__ float fast_exp2(float x) {
#if __has_builtin(__builtin_amdgcn_exp2f)
  return __builtin_amdgcn_exp2f(x);
#else
  return exp2f(x);
#endif
}
__device__ __forceinline__ float fast_rcp(float x) {
#if __has_builtin(__builtin_amdgcn_rcpf)
  return __builtin_amdgcn_rcpf(x);
#else
  return 1.0f / x;
#endif
}

// ---------------------------------------------------------------------------
// k1: h = x @ W^T  (bf16 WMMA, f32 acc). Output stored TRANSPOSED as bf16:
// hT[f][n], f in [0,256), n in [0,6144). One 16x16 tile per wave.
// A-frag (16x32 bf16): lane<16 -> M=lane, K = {0..7, 16..23}; lane>=16 -> K+8.
// B-frag (32x16 bf16): lane%16 = N col, same K order as A (lane>=16 -> K+8).
// ---------------------------------------------------------------------------
__global__ __launch_bounds__(256) void k1_proj(const float* __restrict__ x,
                                               const float* __restrict__ W,
                                               __bf16* __restrict__ hT) {
  const int lane = threadIdx.x & 31;
  const int wave = blockIdx.x * 8 + (threadIdx.x >> 5);   // 6144 waves total
  const int Mi = lane & 15, hi = lane >> 4;
  const int tm = wave >> 4;          // 0..383  (16 rows each)
  const int tn = wave & 15;          // 0..15   (16 out-features each)
  const int I0 = tm << 4, n0 = tn << 4;

  v8f acc = {0.f, 0.f, 0.f, 0.f, 0.f, 0.f, 0.f, 0.f};
  const float* xr = x + (size_t)(I0 + Mi) * INF_;
  const float* wr = W + (size_t)(n0 + Mi) * INF_;

  for (int kk = 0; kk < INF_; kk += 32) {
    const float* ap = xr + kk + hi * 8;
    float4 a0 = *(const float4*)(ap);
    float4 a1 = *(const float4*)(ap + 4);
    float4 a2 = *(const float4*)(ap + 16);
    float4 a3 = *(const float4*)(ap + 20);
    const float* bp = wr + kk + hi * 8;
    float4 b0 = *(const float4*)(bp);
    float4 b1 = *(const float4*)(bp + 4);
    float4 b2 = *(const float4*)(bp + 16);
    float4 b3 = *(const float4*)(bp + 20);

    v16bf ua, ub;
    ua[0]  = (__bf16)a0.x; ua[1]  = (__bf16)a0.y;
    ua[2]  = (__bf16)a0.z; ua[3]  = (__bf16)a0.w;
    ua[4]  = (__bf16)a1.x; ua[5]  = (__bf16)a1.y;
    ua[6]  = (__bf16)a1.z; ua[7]  = (__bf16)a1.w;
    ua[8]  = (__bf16)a2.x; ua[9]  = (__bf16)a2.y;
    ua[10] = (__bf16)a2.z; ua[11] = (__bf16)a2.w;
    ua[12] = (__bf16)a3.x; ua[13] = (__bf16)a3.y;
    ua[14] = (__bf16)a3.z; ua[15] = (__bf16)a3.w;
    ub[0]  = (__bf16)b0.x; ub[1]  = (__bf16)b0.y;
    ub[2]  = (__bf16)b0.z; ub[3]  = (__bf16)b0.w;
    ub[4]  = (__bf16)b1.x; ub[5]  = (__bf16)b1.y;
    ub[6]  = (__bf16)b1.z; ub[7]  = (__bf16)b1.w;
    ub[8]  = (__bf16)b2.x; ub[9]  = (__bf16)b2.y;
    ub[10] = (__bf16)b2.z; ub[11] = (__bf16)b2.w;
    ub[12] = (__bf16)b3.x; ub[13] = (__bf16)b3.y;
    ub[14] = (__bf16)b3.z; ub[15] = (__bf16)b3.w;

    acc = __builtin_amdgcn_wmma_f32_16x16x32_bf16(
        false, ua, false, ub, (short)0, acc, false, false);
  }
  // C layout: VGPR r -> (row = r + 8*hi, col = Mi). Lane holds 8 consecutive
  // rows of fixed feature column -> one contiguous b128 store into hT.
  v8bf st;
  st[0] = (__bf16)acc[0]; st[1] = (__bf16)acc[1];
  st[2] = (__bf16)acc[2]; st[3] = (__bf16)acc[3];
  st[4] = (__bf16)acc[4]; st[5] = (__bf16)acc[5];
  st[6] = (__bf16)acc[6]; st[7] = (__bf16)acc[7];
  *(v8bf*)(hT + (size_t)(n0 + Mi) * Nn + I0 + hi * 8) = st;
}

// ---------------------------------------------------------------------------
// k2: score vectors, PRE-SCALED by log2(e) so k4's inner loop is a raw exp2.
// ssrcT[h][n] = L2E * sum_f h[n,h,f]*a_src[h,f]; same for dst.
// (leaky-ReLU commutes with multiplication by a positive constant.)
// ---------------------------------------------------------------------------
__global__ __launch_bounds__(256) void k2_scores(const __bf16* __restrict__ hT,
                                                 const float* __restrict__ a,
                                                 float* __restrict__ ssrcT,
                                                 float* __restrict__ sdstT) {
  const int n = blockIdx.x * 256 + threadIdx.x;
#pragma unroll
  for (int h = 0; h < HEADS; ++h) {
    float ss = 0.f, sd = 0.f;
#pragma unroll 4
    for (int f = 0; f < FPH; ++f) {
      float hv = (float)hT[(size_t)(h * FPH + f) * Nn + n];
      ss += hv * a[h * (2 * FPH) + f];
      sd += hv * a[h * (2 * FPH) + FPH + f];
    }
    ssrcT[h * Nn + n] = L2E * ss;
    sdstT[h * Nn + n] = L2E * sd;
  }
}

// ---------------------------------------------------------------------------
// k4: fused mask + softmax + aggregation in ONE adj pass.
// Wave = (16 i-rows, 1 head). Block = 8 waves = 2 i-tiles x 4 heads.
// p_unnorm = exp2( lrelu( mask ? ssrc'+sdst' : -inf ) )  built directly in the
// bf16 A-fragment lane layout. 4 value WMMAs + 1 ones-WMMA (row-sum /
// softmax denominator) per 32-j step. Normalize in epilogue: the ones-WMMA
// accumulator holds l_row in VGPR r for exactly the rows this lane stores.
// ---------------------------------------------------------------------------
__device__ __forceinline__ float pfun(int av, float sdv, float ssrc) {
  float t = ssrc + sdv;                       // already scaled by log2(e)
  t = (av > 0) ? t : -__builtin_inff();       // mask BEFORE lrelu: exp2(-inf)=0
  float e = fmaxf(t, NEG_SLOPE * t);          // leaky relu = max(t, 0.2t)
  return fast_exp2(e);
}

__global__ __launch_bounds__(256) void k4_agg(const int* __restrict__ adj,
                                              const __bf16* __restrict__ hT,
                                              const float* __restrict__ ssrcT,
                                              const float* __restrict__ sdstT,
                                              float* __restrict__ out) {
  const int tid = threadIdx.x, lane = tid & 31, wave = tid >> 5;
  const int head = wave & 3, itile = wave >> 2;
  const int I0 = blockIdx.x * 32 + itile * 16;
  const int Mi = lane & 15, hi = lane >> 4, kb = hi * 8;
  const int row = I0 + Mi;

  const float ssrc = ssrcT[head * Nn + row];
  const int* arow = adj + (size_t)row * Nn;
  const float* srow = sdstT + (size_t)head * Nn;
  const __bf16* b0p = hT + (size_t)(head * FPH + 0  + Mi) * Nn + hi * 16;
  const __bf16* b1p = hT + (size_t)(head * FPH + 16 + Mi) * Nn + hi * 16;
  const __bf16* b2p = hT + (size_t)(head * FPH + 32 + Mi) * Nn + hi * 16;
  const __bf16* b3p = hT + (size_t)(head * FPH + 48 + Mi) * Nn + hi * 16;

  const v8u uones = {0x3F803F80u, 0x3F803F80u, 0x3F803F80u, 0x3F803F80u,
                     0x3F803F80u, 0x3F803F80u, 0x3F803F80u, 0x3F803F80u};
  const v16bf ONES = __builtin_bit_cast(v16bf, uones);

  v8f acc0 = {0.f,0.f,0.f,0.f,0.f,0.f,0.f,0.f};
  v8f acc1 = acc0, acc2 = acc0, acc3 = acc0, accL = acc0;

  for (int j0 = 0; j0 < Nn; j0 += 32) {
    int4 a0 = *(const int4*)(arow + j0 + 2 * kb);
    int4 a1 = *(const int4*)(arow + j0 + 2 * kb + 4);
    int4 a2 = *(const int4*)(arow + j0 + 2 * kb + 8);
    int4 a3 = *(const int4*)(arow + j0 + 2 * kb + 12);
    float4 s0 = *(const float4*)(srow + j0 + 2 * kb);
    float4 s1 = *(const float4*)(srow + j0 + 2 * kb + 4);
    float4 s2 = *(const float4*)(srow + j0 + 2 * kb + 8);
    float4 s3 = *(const float4*)(srow + j0 + 2 * kb + 12);
    __builtin_prefetch(arow + j0 + 64, 0, 1);   // global_prefetch_b8 on adj stream

    v16bf A;
    A[0]  = (__bf16)pfun(a0.x, s0.x, ssrc);
    A[1]  = (__bf16)pfun(a0.y, s0.y, ssrc);
    A[2]  = (__bf16)pfun(a0.z, s0.z, ssrc);
    A[3]  = (__bf16)pfun(a0.w, s0.w, ssrc);
    A[4]  = (__bf16)pfun(a1.x, s1.x, ssrc);
    A[5]  = (__bf16)pfun(a1.y, s1.y, ssrc);
    A[6]  = (__bf16)pfun(a1.z, s1.z, ssrc);
    A[7]  = (__bf16)pfun(a1.w, s1.w, ssrc);
    A[8]  = (__bf16)pfun(a2.x, s2.x, ssrc);
    A[9]  = (__bf16)pfun(a2.y, s2.y, ssrc);
    A[10] = (__bf16)pfun(a2.z, s2.z, ssrc);
    A[11] = (__bf16)pfun(a2.w, s2.w, ssrc);
    A[12] = (__bf16)pfun(a3.x, s3.x, ssrc);
    A[13] = (__bf16)pfun(a3.y, s3.y, ssrc);
    A[14] = (__bf16)pfun(a3.z, s3.z, ssrc);
    A[15] = (__bf16)pfun(a3.w, s3.w, ssrc);

    v16bf u0 = *(const v16bf*)(b0p + j0);
    v16bf u1 = *(const v16bf*)(b1p + j0);
    v16bf u2 = *(const v16bf*)(b2p + j0);
    v16bf u3 = *(const v16bf*)(b3p + j0);

    acc0 = __builtin_amdgcn_wmma_f32_16x16x32_bf16(false, A, false, u0,
                                                   (short)0, acc0, false, false);
    acc1 = __builtin_amdgcn_wmma_f32_16x16x32_bf16(false, A, false, u1,
                                                   (short)0, acc1, false, false);
    acc2 = __builtin_amdgcn_wmma_f32_16x16x32_bf16(false, A, false, u2,
                                                   (short)0, acc2, false, false);
    acc3 = __builtin_amdgcn_wmma_f32_16x16x32_bf16(false, A, false, u3,
                                                   (short)0, acc3, false, false);
    accL = __builtin_amdgcn_wmma_f32_16x16x32_bf16(false, A, false, ONES,
                                                   (short)0, accL, false, false);
  }

  const int orow0 = I0 + hi * 8;
#pragma unroll
  for (int r = 0; r < 8; ++r) {
    const float rinv = fast_rcp(accL[r]);     // softmax denominator for row r
    float* op = out + (size_t)(orow0 + r) * OUTF + head * FPH + Mi;
    op[0]  = acc0[r] * rinv;
    op[16] = acc1[r] * rinv;
    op[32] = acc2[r] * rinv;
    op[48] = acc3[r] * rinv;
  }
}

// ---------------------------------------------------------------------------
extern "C" void kernel_launch(void* const* d_in, const int* in_sizes, int n_in,
                              void* d_out, int out_size, void* d_ws, size_t ws_size,
                              hipStream_t stream) {
  (void)in_sizes; (void)n_in; (void)out_size; (void)ws_size;
  const float* x   = (const float*)d_in[0];
  const int*   adj = (const int*)d_in[1];
  const float* W   = (const float*)d_in[2];
  const float* a   = (const float*)d_in[3];
  float* out = (float*)d_out;

  // workspace layout (~3.35 MB total)
  char* ws = (char*)d_ws;
  __bf16* hT   = (__bf16*)(ws);                    // 256*6144*2 = 3145728
  float* ssrcT = (float*)(ws + 3145728);           // 4*6144*4   = 98304
  float* sdstT = (float*)(ws + 3145728 + 98304);   // 98304

  k1_proj<<<768, 256, 0, stream>>>(x, W, hT);
  k2_scores<<<Nn / 256, 256, 0, stream>>>(hT, a, ssrcT, sdstT);
  k4_agg<<<Nn / 32, 256, 0, stream>>>(adj, hT, ssrcT, sdstT, out);
}
